// DeepseekOCRAttention_59167469470468
// MI455X (gfx1250) — compile-verified, matrix operand order checked by reading, not run on hardware
//
#include <hip/hip_runtime.h>
#include <hip/hip_bf16.h>

// ---------------------------------------------------------------------------
// DeepseekOCR attention for gfx1250 (MI455X)
//   - f16 WMMA GEMMs (BM=BN=128, 8 accumulators/wave)
//   - flash attention with async global->LDS staging (ASYNCcnt)
//   B=2, T=2048, HIDDEN=1280, H=10, d=128
// ---------------------------------------------------------------------------

typedef __attribute__((ext_vector_type(16))) _Float16 v16h;
typedef __attribute__((ext_vector_type(8)))  _Float16 v8h;
typedef __attribute__((ext_vector_type(8)))  float    v8f;

#define WMMA_F16(a, b, c) \
    __builtin_amdgcn_wmma_f32_16x16x32_f16(false, (a), false, (b), (short)0, (c), false, false)

static constexpr int BATCH   = 2;
static constexpr int SEQ     = 2048;
static constexpr int HID     = 1280;
static constexpr int NHEAD   = 10;
static constexpr int HDIM    = 128;
static constexpr int ROWS    = BATCH * SEQ;      // 4096
static constexpr float SCALE = 0.08838834764831845f; // 1/sqrt(128)

// Load one 16x32 f16 fragment row (A or column-major B).
// Layout per ISA 7.12.2 (16-bit A matrix 16x32):
//   lanes 0-15 : elems 0..7 -> K=0..7,  elems 8..15 -> K=16..23
//   lanes 16-31: elems 0..7 -> K=8..15, elems 8..15 -> K=24..31
__device__ __forceinline__ v16h ldfrag(const _Float16* rowp, int hs) {
    v8h lo = *reinterpret_cast<const v8h*>(rowp + hs * 8);
    v8h hi = *reinterpret_cast<const v8h*>(rowp + 16 + hs * 8);
    return __builtin_shufflevector(lo, hi, 0, 1, 2, 3, 4, 5, 6, 7,
                                           8, 9, 10, 11, 12, 13, 14, 15);
}

// Async copy of one 16-byte chunk: global (saddr + 32-bit voffset) -> LDS.
// Tracked by ASYNCcnt; drain with s_wait_asynccnt before the barrier.
__device__ __forceinline__ void async_g2l_b128(void* ldsPtr, unsigned gOffBytes,
                                               const void* gBase) {
    unsigned ldsOff = (unsigned)(unsigned long long)(size_t)ldsPtr;
    asm volatile("global_load_async_to_lds_b128 %0, %1, %2"
                 :: "v"(ldsOff), "v"(gOffBytes),
                    "s"((unsigned long long)(size_t)gBase)
                 : "memory");
}
__device__ __forceinline__ void wait_async0() {
    asm volatile("s_wait_asynccnt 0" ::: "memory");
}

// ---------------------------------------------------------------------------
// Tiled WMMA GEMM:  C[M,N] = A[M,K] * B[K,N]
//   BM=BN=128, BK=32; 256 threads = 8 waves; wave owns 32x64 of C
//   (2 A-frags + 4 B-frags -> 8 WMMAs per k-step per wave).
// ---------------------------------------------------------------------------
template <int A_HALF, int OUT_HALF>
__global__ void gemm_wmma_kernel(const void* __restrict__ Ap,
                                 const float* __restrict__ Bp,
                                 void* __restrict__ Cp,
                                 int M, int N, int K) {
    constexpr int BM = 128, BN = 128, BK = 32;
    __shared__ _Float16 As[BM][BK];   // row-major           (8 KB)
    __shared__ _Float16 Bs[BN][BK];   // column-major B[k][n] (8 KB)

    const int tid  = threadIdx.x;
    const int wave = tid >> 5;
    const int lane = tid & 31;
    const int lm   = lane & 15;
    const int hs   = lane >> 4;

    const int m0 = blockIdx.x * BM;
    const int n0 = blockIdx.y * BN;
    const int wm = (wave & 3) * 32;        // wave row offset: two 16-row tiles
    const int wn = (wave >> 2) * 64;       // wave col offset: four 16-col tiles

    v8f acc[2][4];
#pragma unroll
    for (int mi = 0; mi < 2; ++mi)
#pragma unroll
        for (int ni = 0; ni < 4; ++ni)
            acc[mi][ni] = v8f{};

    for (int k0 = 0; k0 < K; k0 += BK) {
        const bool more = (k0 + BK) < K;
        // stage A (fp32->fp16 convert, or fp16 copy)
        for (int i = tid; i < BM * BK; i += 256) {
            int r = i >> 5, c = i & 31;
            float v;
            if (A_HALF)
                v = (float)((const _Float16*)Ap)[(size_t)(m0 + r) * K + k0 + c];
            else
                v = ((const float*)Ap)[(size_t)(m0 + r) * K + k0 + c];
            As[r][c] = (_Float16)v;
            if (more && c == 0) {          // prefetch next A tile row -> global_prefetch_b8
                if (A_HALF)
                    __builtin_prefetch(&((const _Float16*)Ap)[(size_t)(m0 + r) * K + k0 + BK], 0, 1);
                else
                    __builtin_prefetch(&((const float*)Ap)[(size_t)(m0 + r) * K + k0 + BK], 0, 1);
            }
        }
        // stage B transposed to column-major
        for (int i = tid; i < BN * BK; i += 256) {
            int kk = i >> 7, c = i & 127;
            Bs[c][kk] = (_Float16)Bp[(size_t)(k0 + kk) * N + n0 + c];
            if (more && c == 0)            // prefetch next B tile row
                __builtin_prefetch(&Bp[(size_t)(k0 + BK + kk) * N + n0], 0, 1);
        }
        __syncthreads();

        v16h a0 = ldfrag(&As[wm + lm][0], hs);
        v16h a1 = ldfrag(&As[wm + 16 + lm][0], hs);
#pragma unroll
        for (int ni = 0; ni < 4; ++ni) {
            v16h b = ldfrag(&Bs[wn + ni * 16 + lm][0], hs);
            acc[0][ni] = WMMA_F16(a0, b, acc[0][ni]);
            acc[1][ni] = WMMA_F16(a1, b, acc[1][ni]);
        }
        __syncthreads();
    }

    // C layout: VGPR r -> row (r + hs*8), col lm
#pragma unroll
    for (int mi = 0; mi < 2; ++mi) {
#pragma unroll
        for (int ni = 0; ni < 4; ++ni) {
#pragma unroll
            for (int r = 0; r < 8; ++r) {
                int m = m0 + wm + mi * 16 + r + hs * 8;
                int n = n0 + wn + ni * 16 + lm;
                if (OUT_HALF)
                    ((_Float16*)Cp)[(size_t)m * N + n] = (_Float16)acc[mi][ni][r];
                else
                    ((float*)Cp)[(size_t)m * N + n] = acc[mi][ni][r];
            }
        }
    }
}

// ---------------------------------------------------------------------------
// RoPE applied in-place to f16 Q and K, layout [ROWS][HID], head-major cols.
// ---------------------------------------------------------------------------
__global__ void rope_kernel(_Float16* __restrict__ q, _Float16* __restrict__ k) {
    int idx = blockIdx.x * blockDim.x + threadIdx.x;     // over ROWS*NHEAD*64
    if (idx >= ROWS * NHEAD * 64) return;
    int row = idx / (NHEAD * 64);
    int rem = idx - row * (NHEAD * 64);
    int h = rem >> 6;
    int i = rem & 63;
    int t = row & (SEQ - 1);                             // position within sequence

    // inv_freq = theta^(-2i/128) = exp(-i * ln(1e4)/64)
    float ang = (float)t * __expf(-(float)i * 0.14391156f);
    float s, c;
    __sincosf(ang, &s, &c);

    size_t base = (size_t)row * HID + h * HDIM + i;
    float q1 = (float)q[base], q2 = (float)q[base + 64];
    float k1 = (float)k[base], k2 = (float)k[base + 64];
    q[base]      = (_Float16)(q1 * c - q2 * s);
    q[base + 64] = (_Float16)(q2 * c + q1 * s);
    k[base]      = (_Float16)(k1 * c - k2 * s);
    k[base + 64] = (_Float16)(k2 * c + k1 * s);
}

// ---------------------------------------------------------------------------
// Flash-style causal attention, one workgroup per (64-query-row block, b*h).
// Q,K tiles staged with async global->LDS; V transposed manually.
// ---------------------------------------------------------------------------
__global__ void attn_kernel(const _Float16* __restrict__ Q,
                            const _Float16* __restrict__ K,
                            const _Float16* __restrict__ V,
                            _Float16* __restrict__ O) {
    __shared__ _Float16 Qs[64][HDIM];   // 16 KB
    __shared__ _Float16 Ks[64][HDIM];   // 16 KB (row-major == col-major B for QK^T)
    __shared__ _Float16 Vt[HDIM][64];   // 16 KB (transposed: col-major B for P*V)
    __shared__ float    Sb[64][64];     // 16 KB
    __shared__ _Float16 Pb[64][64];     // 8 KB
    __shared__ float    rowM[64], rowL[64], rowC[64];

    const int tid  = threadIdx.x;
    const int wave = tid >> 5;
    const int lane = tid & 31;
    const int lm   = lane & 15;
    const int hs   = lane >> 4;

    const int i0 = blockIdx.x * 64;          // query block base (within sequence)
    const int bh = blockIdx.y;
    const int b  = bh / NHEAD;
    const int h  = bh - b * NHEAD;
    const size_t rowBase = (size_t)b * SEQ;
    const size_t colBase = (size_t)h * HDIM;

    const int wm = (wave & 3) * 16;          // wave's S/O row offset
    const int wn = (wave >> 2) * 32;         // wave's S col offset (two tiles)
    const int nO = (wave >> 2) * 64;         // wave's O col offset (four tiles)

    // async-stage Q block: 64 rows x 256 B = 1024 x b128 chunks
    for (int c = tid; c < 1024; c += 256) {
        int r = c >> 4, col = (c & 15) << 3;               // col in halves
        unsigned gOff = (unsigned)(((rowBase + i0 + r) * HID + colBase + col) * 2);
        async_g2l_b128(&Qs[r][col], gOff, Q);
    }
    if (tid < 64) { rowM[tid] = -1e30f; rowL[tid] = 0.0f; }

    v8f o0 = {}, o1 = {}, o2 = {}, o3 = {};

    for (int j0 = 0; j0 <= i0; j0 += 64) {
        // async-stage K block
        for (int c = tid; c < 1024; c += 256) {
            int r = c >> 4, col = (c & 15) << 3;
            unsigned gOff = (unsigned)(((rowBase + j0 + r) * HID + colBase + col) * 2);
            async_g2l_b128(&Ks[r][col], gOff, K);
        }
        // stage V transposed (needs shuffle; keep VALU path)
        for (int i = tid; i < 64 * HDIM; i += 256) {
            int r = i >> 7, c = i & 127;
            Vt[c][r] = V[(rowBase + j0 + r) * HID + colBase + c];
        }
        wait_async0();
        __syncthreads();

        // S = Q * K^T over d=128 (4 WMMA k-steps), two 16x16 tiles per wave
        v8f s0 = {}, s1 = {};
#pragma unroll
        for (int kk = 0; kk < HDIM; kk += 32) {
            v16h a  = ldfrag(&Qs[wm + lm][kk], hs);
            v16h b0 = ldfrag(&Ks[wn + lm][kk], hs);
            v16h b1 = ldfrag(&Ks[wn + 16 + lm][kk], hs);
            s0 = WMMA_F16(a, b0, s0);
            s1 = WMMA_F16(a, b1, s1);
        }
#pragma unroll
        for (int r = 0; r < 8; ++r) {
            Sb[wm + r + hs * 8][wn + lm]      = s0[r];
            Sb[wm + r + hs * 8][wn + 16 + lm] = s1[r];
        }
        __syncthreads();

        // online softmax over this 64-wide key block (threads 0..63, one row each)
        if (tid < 64) {
            int row = tid;
            int qi  = i0 + row;
            int nv  = qi - j0 + 1; if (nv > 64) nv = 64;   // causal validity
            float mold = rowM[row];
            float mx = mold;
            for (int n = 0; n < nv; ++n) mx = fmaxf(mx, Sb[row][n] * SCALE);
            float corr = __expf(mold - mx);
            float l = rowL[row] * corr;
            for (int n = 0; n < 64; ++n) {
                float p = (n < nv) ? __expf(Sb[row][n] * SCALE - mx) : 0.0f;
                Pb[row][n] = (_Float16)p;
                l += p;
            }
            rowM[row] = mx; rowL[row] = l; rowC[row] = corr;
        }
        __syncthreads();

        // rescale running accumulators by per-row correction
#pragma unroll
        for (int r = 0; r < 8; ++r) {
            float f = rowC[wm + r + hs * 8];
            o0[r] *= f; o1[r] *= f; o2[r] *= f; o3[r] *= f;
        }
        // O += P (64x64) * V (64x128): two WMMA k-steps, four n-tiles per wave
#pragma unroll
        for (int kk = 0; kk < 64; kk += 32) {
            v16h a  = ldfrag(&Pb[wm + lm][kk], hs);
            v16h b0 = ldfrag(&Vt[nO + lm][kk], hs);
            v16h b1 = ldfrag(&Vt[nO + 16 + lm][kk], hs);
            v16h b2 = ldfrag(&Vt[nO + 32 + lm][kk], hs);
            v16h b3 = ldfrag(&Vt[nO + 48 + lm][kk], hs);
            o0 = WMMA_F16(a, b0, o0);
            o1 = WMMA_F16(a, b1, o1);
            o2 = WMMA_F16(a, b2, o2);
            o3 = WMMA_F16(a, b3, o3);
        }
        __syncthreads();
    }

    // epilogue: divide by row sum, store f16
#pragma unroll
    for (int r = 0; r < 8; ++r) {
        int lrow = wm + r + hs * 8;
        float inv = 1.0f / rowL[lrow];
        size_t base = (rowBase + i0 + lrow) * HID + colBase + nO + lm;
        O[base]      = (_Float16)(o0[r] * inv);
        O[base + 16] = (_Float16)(o1[r] * inv);
        O[base + 32] = (_Float16)(o2[r] * inv);
        O[base + 48] = (_Float16)(o3[r] * inv);
    }
}

// ---------------------------------------------------------------------------
// Host-side launch
// ---------------------------------------------------------------------------
extern "C" void kernel_launch(void* const* d_in, const int* in_sizes, int n_in,
                              void* d_out, int out_size, void* d_ws, size_t ws_size,
                              hipStream_t stream) {
    (void)in_sizes; (void)n_in; (void)out_size; (void)ws_size;

    const float* X  = (const float*)d_in[0];   // [2,2048,1280] -> [4096][1280]
    const float* Wq = (const float*)d_in[1];   // [1280][1280]
    const float* Wk = (const float*)d_in[2];
    const float* Wv = (const float*)d_in[3];
    const float* Wo = (const float*)d_in[4];
    float* out = (float*)d_out;                // [4096][1280]

    const size_t elems = (size_t)ROWS * HID;   // 10 MB per f16 buffer
    _Float16* q  = (_Float16*)d_ws;
    _Float16* k  = q + elems;
    _Float16* v  = k + elems;
    _Float16* ao = v + elems;                  // 40 MB total workspace

    dim3 gGemm(ROWS / 128, HID / 128);         // (32, 10)

    // QKV projections (fp32 in, f16 out)
    gemm_wmma_kernel<0, 1><<<gGemm, 256, 0, stream>>>(X, Wq, q, ROWS, HID, HID);
    gemm_wmma_kernel<0, 1><<<gGemm, 256, 0, stream>>>(X, Wk, k, ROWS, HID, HID);
    gemm_wmma_kernel<0, 1><<<gGemm, 256, 0, stream>>>(X, Wv, v, ROWS, HID, HID);

    // RoPE on Q,K
    int nrope = ROWS * NHEAD * 64;
    rope_kernel<<<(nrope + 255) / 256, 256, 0, stream>>>(q, k);

    // causal flash attention
    attn_kernel<<<dim3(SEQ / 64, BATCH * NHEAD), 256, 0, stream>>>(q, k, v, ao);

    // output projection (f16 A, fp32 out)
    gemm_wmma_kernel<1, 0><<<gGemm, 256, 0, stream>>>(ao, Wo, out, ROWS, HID, HID);
}
